// StreamingTransformer_44710609551805
// MI455X (gfx1250) — matrix operand order repaired and checked by reading, not compiled, not run on hardware
//
#include <hip/hip_runtime.h>
#include <hip/hip_bf16.h>

#define DEVINL __device__ __forceinline__

typedef __attribute__((ext_vector_type(16))) _Float16 v16h;
typedef __attribute__((ext_vector_type(8)))  _Float16 v8h;
typedef __attribute__((ext_vector_type(4)))  _Float16 v4h;
typedef __attribute__((ext_vector_type(8)))  float    v8f;
typedef __attribute__((ext_vector_type(4)))  int      v4i;

constexpr int Bn = 4, Sn = 2048, Cn = 1024, Hn = 16, Dn = 64, HISTn = 400;
constexpr int Mtok = Bn * Sn;  // 8192

// ---------------------------------------------------------------- async path
#if defined(__has_builtin)
#if __has_builtin(__builtin_amdgcn_global_load_async_to_lds_b128)
#define USE_ASYNC_BUILTIN 1
#endif
#if __has_builtin(__builtin_amdgcn_s_wait_asynccnt)
#define USE_WAIT_BUILTIN 1
#endif
#endif

typedef __attribute__((address_space(1))) v4i gv4i;  // global int4
typedef __attribute__((address_space(3))) v4i lv4i;  // LDS int4

// Copy 16 bytes global -> LDS, asynchronously (ASYNCcnt-tracked).
DEVINL void async_cp16(const _Float16* g, _Float16* l) {
#if defined(USE_ASYNC_BUILTIN)
  __builtin_amdgcn_global_load_async_to_lds_b128((gv4i*)g, (lv4i*)l, 0, 0);
#else
  // VDST = LDS byte address (low 32 bits of the flat shared address),
  // VADDR = 64-bit global address.
  unsigned lofs = (unsigned)(size_t)l;
  asm volatile("global_load_async_to_lds_b128 %0, %1, off" ::"v"(lofs), "v"(g)
               : "memory");
#endif
}

template <int N>
DEVINL void wait_async() {
#if defined(USE_WAIT_BUILTIN)
  __builtin_amdgcn_s_wait_asynccnt((short)N);
#else
  if constexpr (N == 0)
    asm volatile("s_wait_asynccnt 0x0" ::: "memory");
  else
    asm volatile("s_wait_asynccnt 0x4" ::: "memory");
#endif
}

// ---------------------------------------------------------------- wmma utils
DEVINL v8f vzero() {
  v8f z;
#pragma unroll
  for (int i = 0; i < 8; ++i) z[i] = 0.f;
  return z;
}

DEVINL v8f wmma_f16(v16h a, v16h b, v8f c) {
  return __builtin_amdgcn_wmma_f32_16x16x32_f16(false, a, false, b, (short)0, c,
                                                false, false);
}

// A-fragment (16 rows x 32 k, f16). Memory: row-major, row stride ld halves.
// Lane L: row = L%16, K in {koff..koff+7} U {koff+16..koff+23}, koff = 8*(L/16).
DEVINL v16h load_a_frag(const _Float16* p, int ld) {
  const int lane = threadIdx.x & 31;
  const _Float16* r = p + (size_t)(lane & 15) * ld + ((lane >> 4) << 3);
  v8h lo = *(const v8h*)r;
  v8h hi = *(const v8h*)(r + 16);
  v16h o;
#pragma unroll
  for (int i = 0; i < 8; ++i) { o[i] = lo[i]; o[8 + i] = hi[i]; }
  return o;
}

// B-fragment (32 k x 16 cols, f16). Memory organized [col][k] (col stride ld).
// Lane L: col = L%16, K in {koff..koff+7} U {koff+16..koff+23}, koff = 8*(L/16).
DEVINL v16h load_b_frag(const _Float16* p, int ld) {
  const int lane = threadIdx.x & 31;
  const _Float16* r = p + (size_t)(lane & 15) * ld + ((lane >> 4) << 3);
  v8h lo = *(const v8h*)r;
  v8h hi = *(const v8h*)(r + 16);
  v16h o;
#pragma unroll
  for (int i = 0; i < 8; ++i) { o[i] = lo[i]; o[8 + i] = hi[i]; }
  return o;
}

// ---------------------------------------------------------------- converts
__global__ __launch_bounds__(256) void cvt_f32_f16_kernel(
    const float* __restrict__ s, _Float16* __restrict__ d, int n4) {
  int i = blockIdx.x * 256 + threadIdx.x;
  if (i < n4) {
    float4 v = ((const float4*)s)[i];
    v4h o;
    o[0] = (_Float16)v.x; o[1] = (_Float16)v.y;
    o[2] = (_Float16)v.z; o[3] = (_Float16)v.w;
    ((v4h*)d)[i] = o;
  }
}

// ---------------------------------------------------------------- layernorm
__global__ __launch_bounds__(256) void ln_kernel(const float* __restrict__ x,
                                                 const float* __restrict__ g,
                                                 const float* __restrict__ bb,
                                                 _Float16* __restrict__ xn) {
  const int lane = threadIdx.x & 31;
  const int w = threadIdx.x >> 5;
  const size_t tok = (size_t)blockIdx.x * 8 + w;  // one wave per token
  const float* row = x + tok * Cn;
  float4 vb[8];
  float sum = 0.f, sq = 0.f;
#pragma unroll
  for (int i = 0; i < 8; ++i) {
    float4 v = ((const float4*)row)[lane * 8 + i];
    vb[i] = v;
    sum += v.x + v.y + v.z + v.w;
    sq += v.x * v.x + v.y * v.y + v.z * v.z + v.w * v.w;
  }
#pragma unroll
  for (int mm = 1; mm < 32; mm <<= 1) {
    sum += __shfl_xor(sum, mm, 32);
    sq += __shfl_xor(sq, mm, 32);
  }
  const float mu = sum * (1.f / Cn);
  const float rs = rsqrtf(sq * (1.f / Cn) - mu * mu + 1e-5f);
  _Float16* orow = xn + tok * Cn;
#pragma unroll
  for (int i = 0; i < 8; ++i) {
    float4 gv = ((const float4*)g)[lane * 8 + i];
    float4 bv = ((const float4*)bb)[lane * 8 + i];
    v4h ov;
    ov[0] = (_Float16)((vb[i].x - mu) * rs * gv.x + bv.x);
    ov[1] = (_Float16)((vb[i].y - mu) * rs * gv.y + bv.y);
    ov[2] = (_Float16)((vb[i].z - mu) * rs * gv.z + bv.z);
    ov[3] = (_Float16)((vb[i].w - mu) * rs * gv.w + bv.w);
    *(v4h*)(orow + (lane * 8 + i) * 4) = ov;
  }
}

// ---------------------------------------------------------------- WMMA GEMM
// Y[m,n] = sum_k A[m,k] * W[n,k] + bias[n]
// Async double-buffered LDS pipeline: 128x128 block tile, k-step 32.
// MODE 0: f16 out, row-major [M,N]
// MODE 1: f16 out, transposed per-head layout Vt[B,H,D,S] (for V)
// MODE 2: f32 out, (acc + bias) * lscale[n]  (final projection)
template <int MODE>
__global__ __launch_bounds__(256) void gemm_wmma_kernel(
    const _Float16* __restrict__ A, const _Float16* __restrict__ W,
    const float* __restrict__ bias, const float* __restrict__ lscale,
    void* __restrict__ outp, int M, int N, int K) {
  __shared__ __align__(16) _Float16 At[2][128][40];  // 32 k-halves + 8 pad
  __shared__ __align__(16) _Float16 Bt[2][128][40];

  const int tid = threadIdx.x;
  const int lane = tid & 31;
  const int w = tid >> 5;
  const int mblk = blockIdx.y * 128;
  const int nblk = blockIdx.x * 128;
  const int m_off = (w & 3) * 32;   // wave: 32 rows of the block tile
  const int n_off = (w >> 2) * 64;  // wave: 64 cols of the block tile

  // Staging: tile = 128 rows x 32 halves = 512 x 16B chunks; 256 threads
  // issue 2 chunks each per matrix (4 async ops / thread / k-step).
  const int crow = tid >> 1;
  const int ccol = (tid & 1) * 8;  // halves

  v8f acc[2][4];
#pragma unroll
  for (int i = 0; i < 2; ++i)
#pragma unroll
    for (int j = 0; j < 4; ++j) acc[i][j] = vzero();

  const int NK = K / 32;

  {  // prologue: stage tile 0
    const _Float16* Ag = A + (size_t)(mblk + crow) * K;
    const _Float16* Bg = W + (size_t)(nblk + crow) * K;
    async_cp16(Ag + ccol, &At[0][crow][ccol]);
    async_cp16(Ag + ccol + 16, &At[0][crow][ccol + 16]);
    async_cp16(Bg + ccol, &Bt[0][crow][ccol]);
    async_cp16(Bg + ccol + 16, &Bt[0][crow][ccol + 16]);
  }

  for (int it = 0; it < NK; ++it) {
    if (it + 1 < NK) {  // stage next tile into the other buffer
      const int nb = (it + 1) & 1;
      const int k0 = (it + 1) * 32;
      const _Float16* Ag = A + (size_t)(mblk + crow) * K + k0;
      const _Float16* Bg = W + (size_t)(nblk + crow) * K + k0;
      async_cp16(Ag + ccol, &At[nb][crow][ccol]);
      async_cp16(Ag + ccol + 16, &At[nb][crow][ccol + 16]);
      async_cp16(Bg + ccol, &Bt[nb][crow][ccol]);
      async_cp16(Bg + ccol + 16, &Bt[nb][crow][ccol + 16]);
      wait_async<4>();  // tile `it`'s 4 ops (in-order) have landed
    } else {
      wait_async<0>();
    }
    __syncthreads();  // all waves' async copies for tile `it` visible

    const int buf = it & 1;
    v16h af[2], bf[4];
#pragma unroll
    for (int i = 0; i < 2; ++i)
      af[i] = load_a_frag(&At[buf][m_off + 16 * i][0], 40);
#pragma unroll
    for (int j = 0; j < 4; ++j)
      bf[j] = load_b_frag(&Bt[buf][n_off + 16 * j][0], 40);
#pragma unroll
    for (int i = 0; i < 2; ++i)
#pragma unroll
      for (int j = 0; j < 4; ++j) acc[i][j] = wmma_f16(af[i], bf[j], acc[i][j]);

    __syncthreads();  // done reading buf before it is overwritten
  }

  const int lcol = lane & 15, lgrp = lane >> 4;
#pragma unroll
  for (int i = 0; i < 2; ++i) {
#pragma unroll
    for (int j = 0; j < 4; ++j) {
      const int col = nblk + n_off + 16 * j + lcol;
      const float bval = bias[col];
#pragma unroll
      for (int r = 0; r < 8; ++r) {
        const int row = mblk + m_off + 16 * i + r + 8 * lgrp;
        const float val = acc[i][j][r] + bval;
        if (MODE == 0) {
          ((_Float16*)outp)[(size_t)row * N + col] = (_Float16)val;
        } else if (MODE == 1) {
          const int b = row >> 11, s = row & (Sn - 1);
          const int h = col >> 6, d = col & 63;
          ((_Float16*)outp)[((size_t)((b * Hn + h) * Dn + d)) * Sn + s] =
              (_Float16)val;
        } else {
          ((float*)outp)[(size_t)row * N + col] = val * lscale[col];
        }
      }
    }
  }
}

// ---------------------------------------------------------------- attention
// Flash-style: wave owns 16 query rows; streams keys 64 at a time.
// Q,K: f16 [B,S,C] (head slice stride C). Vt: f16 [B,H,D,S]. O: f16 [B,S,C].
__global__ __launch_bounds__(256) void attn_kernel(
    const _Float16* __restrict__ Q, const _Float16* __restrict__ Km,
    const _Float16* __restrict__ Vt, const float* __restrict__ pe,
    _Float16* __restrict__ O) {
  __shared__ float pe_s[2 * HISTn];
  __shared__ __align__(16) _Float16 P[8][16][72];  // per-wave P tile, padded

  const int lane = threadIdx.x & 31;
  const int w = threadIdx.x >> 5;
  const int b = blockIdx.z, h = blockIdx.y;
  const int q0 = blockIdx.x * 128 + w * 16;
  const int lcol = lane & 15, lgrp = lane >> 4;

  for (int i = threadIdx.x; i < 2 * HISTn; i += 256) pe_s[i] = pe[i];
  __syncthreads();

  const _Float16* Qp = Q + ((size_t)(b * Sn + q0)) * Cn + h * Dn;
  v16h qf[2];
  qf[0] = load_a_frag(Qp, Cn);
  qf[1] = load_a_frag(Qp + 32, Cn);

  v8f o[4];
#pragma unroll
  for (int nt = 0; nt < 4; ++nt) o[nt] = vzero();
  float mrow[8], lrow[8];
#pragma unroll
  for (int r = 0; r < 8; ++r) { mrow[r] = -1e30f; lrow[r] = 0.f; }

  const float scale = 0.125f;  // 1/sqrt(D)

  for (int kt = 0; kt < Sn; kt += 64) {
    // ---- scores: 16 queries x 64 keys
    v8f sc[4];
#pragma unroll
    for (int nt = 0; nt < 4; ++nt) {
      const _Float16* Kp = Km + ((size_t)(b * Sn + kt + nt * 16)) * Cn + h * Dn;
      v16h k0 = load_b_frag(Kp, Cn);
      v16h k1 = load_b_frag(Kp + 32, Cn);
      v8f a = vzero();
      a = wmma_f16(qf[0], k0, a);
      a = wmma_f16(qf[1], k1, a);
      sc[nt] = a;
    }
    // ---- relative-position bias + online softmax
#pragma unroll
    for (int r = 0; r < 8; ++r) {
      const int qi = q0 + r + 8 * lgrp;
      float vals[4];
      float tmax = -1e30f;
#pragma unroll
      for (int nt = 0; nt < 4; ++nt) {
        const int kj = kt + nt * 16 + lcol;
        int rel = qi - kj;
        rel = rel < -HISTn ? -HISTn : rel;
        rel = rel > HISTn - 1 ? HISTn - 1 : rel;
        float vv = sc[nt][r] * scale + pe_s[rel + HISTn];
        vals[nt] = vv;
        tmax = fmaxf(tmax, vv);
      }
#pragma unroll
      for (int mm = 1; mm < 16; mm <<= 1)
        tmax = fmaxf(tmax, __shfl_xor(tmax, mm, 32));
      const float mnew = fmaxf(mrow[r], tmax);
      const float corr = __expf(mrow[r] - mnew);
      mrow[r] = mnew;
      float psum = 0.f;
#pragma unroll
      for (int nt = 0; nt < 4; ++nt) {
        const float pv = __expf(vals[nt] - mnew);
        psum += pv;
        P[w][r + 8 * lgrp][nt * 16 + lcol] = (_Float16)pv;
      }
#pragma unroll
      for (int mm = 1; mm < 16; mm <<= 1) psum += __shfl_xor(psum, mm, 32);
      lrow[r] = lrow[r] * corr + psum;
#pragma unroll
      for (int nt = 0; nt < 4; ++nt) o[nt][r] *= corr;
    }
    // ---- P (from LDS, layout transpose) @ V
    v16h pf0 = load_a_frag(&P[w][0][0], 72);
    v16h pf1 = load_a_frag(&P[w][0][32], 72);
#pragma unroll
    for (int nt = 0; nt < 4; ++nt) {
      const _Float16* Vp =
          Vt + ((size_t)((b * Hn + h) * Dn + nt * 16)) * Sn + kt;
      v16h v0 = load_b_frag(Vp, Sn);
      v16h v1 = load_b_frag(Vp + 32, Sn);
      o[nt] = wmma_f16(pf0, v0, o[nt]);
      o[nt] = wmma_f16(pf1, v1, o[nt]);
    }
  }
  // ---- normalize and store
#pragma unroll
  for (int nt = 0; nt < 4; ++nt) {
#pragma unroll
    for (int r = 0; r < 8; ++r) {
      const int row = q0 + r + 8 * lgrp;
      const int col = h * Dn + nt * 16 + lcol;
      O[((size_t)(b * Sn + row)) * Cn + col] = (_Float16)(o[nt][r] / lrow[r]);
    }
  }
}

// ---------------------------------------------------------------- launcher
extern "C" void kernel_launch(void* const* d_in, const int* in_sizes, int n_in,
                              void* d_out, int out_size, void* d_ws,
                              size_t ws_size, hipStream_t stream) {
  const float* x = (const float*)d_in[0];
  const float* ln_g = (const float*)d_in[1];
  const float* ln_b = (const float*)d_in[2];
  const float* Wq = (const float*)d_in[3];
  const float* bq = (const float*)d_in[4];
  const float* Wk = (const float*)d_in[5];
  const float* bk = (const float*)d_in[6];
  const float* Wv = (const float*)d_in[7];
  const float* bv = (const float*)d_in[8];
  const float* Wo = (const float*)d_in[9];
  const float* bo = (const float*)d_in[10];
  const float* lscale = (const float*)d_in[11];
  const float* pe = (const float*)d_in[12];
  float* out = (float*)d_out;

  char* ws = (char*)d_ws;
  const size_t actBytes = (size_t)Mtok * Cn * sizeof(_Float16);  // 16 MB
  const size_t wBytes = (size_t)Cn * Cn * sizeof(_Float16);      // 2 MB
  _Float16* xn = (_Float16*)ws;            ws += actBytes;
  _Float16* Wqh = (_Float16*)ws;           ws += wBytes;
  _Float16* Wkh = (_Float16*)ws;           ws += wBytes;
  _Float16* Wvh = (_Float16*)ws;           ws += wBytes;
  _Float16* Woh = (_Float16*)ws;           ws += wBytes;
  _Float16* Qh = (_Float16*)ws;            ws += actBytes;
  _Float16* Kh = (_Float16*)ws;            ws += actBytes;
  _Float16* Vth = (_Float16*)ws;           ws += actBytes;
  _Float16* Oh = (_Float16*)ws;            ws += actBytes;

  const int n4 = Cn * Cn / 4;
  const int cvtBlocks = (n4 + 255) / 256;
  cvt_f32_f16_kernel<<<cvtBlocks, 256, 0, stream>>>(Wq, Wqh, n4);
  cvt_f32_f16_kernel<<<cvtBlocks, 256, 0, stream>>>(Wk, Wkh, n4);
  cvt_f32_f16_kernel<<<cvtBlocks, 256, 0, stream>>>(Wv, Wvh, n4);
  cvt_f32_f16_kernel<<<cvtBlocks, 256, 0, stream>>>(Wo, Woh, n4);

  ln_kernel<<<Mtok / 8, 256, 0, stream>>>(x, ln_g, ln_b, xn);

  dim3 ggrid(Cn / 128, Mtok / 128);  // (8, 64)
  gemm_wmma_kernel<0><<<ggrid, 256, 0, stream>>>(xn, Wqh, bq, nullptr, Qh,
                                                 Mtok, Cn, Cn);
  gemm_wmma_kernel<0><<<ggrid, 256, 0, stream>>>(xn, Wkh, bk, nullptr, Kh,
                                                 Mtok, Cn, Cn);
  gemm_wmma_kernel<1><<<ggrid, 256, 0, stream>>>(xn, Wvh, bv, nullptr, Vth,
                                                 Mtok, Cn, Cn);

  dim3 agrid(Sn / 128, Hn, Bn);  // (16, 16, 4)
  attn_kernel<<<agrid, 256, 0, stream>>>(Qh, Kh, Vth, pe, Oh);

  gemm_wmma_kernel<2><<<ggrid, 256, 0, stream>>>(Oh, Woh, bo, lscale, out,
                                                 Mtok, Cn, Cn);
}
